// UTDResNetAdaptiveK_6176162972410
// MI455X (gfx1250) — matrix operand order repaired and checked by reading, not compiled, not run on hardware
//
#include <hip/hip_runtime.h>
#include <hip/hip_bf16.h>
#include <math.h>

// ---------------- types / constants ----------------
typedef unsigned short ushort_t;
typedef __attribute__((ext_vector_type(16))) __bf16 bf16x16;
typedef __attribute__((ext_vector_type(8)))  float  v8f;

#define B_N   8192
#define D_N   384
#define H_N   256
#define C_N   6
#define NSPLIT 8                        // column splits for knn
#define COLS_PER_SPLIT (B_N / NSPLIT)   // 1024
#define NLIST (NSPLIT * 2)              // partial top-k lists per row (2 halves/split)
#define SB_LD 392                       // LDS staging row stride (ushorts), bank-spread
#define ST_LD 37                        // LDS score tile row stride (floats), conflict-free

// ---------------- bf16 helpers ----------------
__device__ inline ushort_t f2bf(float f) {              // round-to-nearest-even
  unsigned u = __float_as_uint(f);
  u += 0x7FFFu + ((u >> 16) & 1u);
  return (ushort_t)(u >> 16);
}
__device__ inline float bf2f(ushort_t h) { return __uint_as_float(((unsigned)h) << 16); }

union Frag { int4 i2[2]; bf16x16 v; };

// A fragment: 16x32 bf16, rows rbase.., K kbase..+31, row-major src[ld]
// lanes<16 hold K {0..15}, lanes>=16 hold K {16..31}
__device__ inline bf16x16 load_a_frag(const ushort_t* __restrict__ src, int ld,
                                      int rbase, int kbase, int lane) {
  int m = lane & 15, hi = lane >> 4;
  const ushort_t* p = src + (size_t)(rbase + m) * ld + kbase + hi * 16;
  Frag f;
  f.i2[0] = *(const int4*)p;
  f.i2[1] = *(const int4*)(p + 8);
  return f.v;
}
// B fragment from global: B[k][n] = srcT[n][k], srcT row-major [N][ld]
__device__ inline bf16x16 load_b_frag(const ushort_t* __restrict__ srcT, int ld,
                                      int cbase, int kbase, int lane) {
  int n = lane & 15, hi = lane >> 4;
  const ushort_t* p = srcT + (size_t)(cbase + n) * ld + kbase + hi * 16;
  Frag f;
  f.i2[0] = *(const int4*)p;
  f.i2[1] = *(const int4*)(p + 8);
  return f.v;
}
// B fragment from LDS staging (row stride SB_LD)
__device__ inline bf16x16 load_b_lds(const ushort_t* sm, int cl, int kbase, int lane) {
  int n = lane & 15, hi = lane >> 4;
  const ushort_t* p = sm + (cl + n) * SB_LD + kbase + hi * 16;
  Frag f;
  f.i2[0] = *(const int4*)p;
  f.i2[1] = *(const int4*)(p + 8);
  return f.v;
}
__device__ inline v8f wmma_bf16(bf16x16 a, bf16x16 b, v8f c) {
  return __builtin_amdgcn_wmma_f32_16x16x32_bf16(false, a, false, b, (short)0, c,
                                                 false, false);
}

// sorted (ascending) top-16 insert, fully unrolled, static indexing only
__device__ inline void topk_insert(float (&bd)[16], int (&bi)[16], float d, int id) {
#pragma unroll
  for (int j = 0; j < 16; ++j) {
    if (d < bd[j]) {
      float td = bd[j]; int ti = bi[j];
      bd[j] = d; bi[j] = id;
      d = td; id = ti;
    }
  }
}

// ---------------- kernel 1: x -> (x_hi,x_lo bf16), sq = ||x||^2, tau -> k ----------
__global__ void prep_x_kernel(const float* __restrict__ x,
                              const float* __restrict__ W_tau,
                              const float* __restrict__ b_tau,
                              ushort_t* __restrict__ x_hi, ushort_t* __restrict__ x_lo,
                              float* __restrict__ sq, int* __restrict__ kk) {
  int w = threadIdx.x >> 5, lane = threadIdx.x & 31;
  int row = blockIdx.x * 8 + w;
  const float* xr = x + (size_t)row * D_N;
  float s = 0.f, tp = 0.f;
#pragma unroll
  for (int t = 0; t < D_N / 32; ++t) {
    int c = lane + t * 32;
    float v = xr[c];
    s += v * v;
    tp += v * W_tau[c];
    ushort_t hb = f2bf(v);
    x_hi[(size_t)row * D_N + c] = hb;
    x_lo[(size_t)row * D_N + c] = f2bf(v - bf2f(hb));
  }
  for (int o = 16; o > 0; o >>= 1) {
    s  += __shfl_down(s, o, 32);
    tp += __shfl_down(tp, o, 32);
  }
  if (lane == 0) {
    sq[row] = s;
    float tau = 1.f / (1.f + expf(-(tp + b_tau[0])));
    int k = (int)rintf(16.f - 12.f * tau);
    kk[row] = k < 1 ? 1 : (k > 16 ? 16 : k);
  }
}

// ---------------- kernel 2: transpose + hi/lo split weight [K,N] -> [N,K] ----------
__global__ void tsplit_kernel(const float* __restrict__ src,
                              ushort_t* __restrict__ dh, ushort_t* __restrict__ dl,
                              int K, int N) {
  int i = blockIdx.x * blockDim.x + threadIdx.x;
  if (i >= K * N) return;
  int k = i / N, n = i - k * N;
  float v = src[i];
  ushort_t hb = f2bf(v);
  dh[(size_t)n * K + k] = hb;
  dl[(size_t)n * K + k] = f2bf(v - bf2f(hb));
}

// ---------------- kernel 3: h = relu(x @ W_proj + b_proj), f32 out -----------------
// min-waves/EU = 2 -> VGPR budget >= 256: A hi+lo (192 VGPRs) stays resident, no spill
__global__ __launch_bounds__(256, 2) void proj_kernel(
    const ushort_t* __restrict__ x_hi, const ushort_t* __restrict__ x_lo,
    const ushort_t* __restrict__ Wpt_hi, const ushort_t* __restrict__ Wpt_lo,
    const float* __restrict__ b_proj, float* __restrict__ h) {
  int w = threadIdx.x >> 5, lane = threadIdx.x & 31;
  int rb = (blockIdx.x * 8 + w) * 16;
  int nc = lane & 15, hic = lane >> 4;
  bf16x16 ahi[D_N / 32], alo[D_N / 32];
#pragma unroll
  for (int ks = 0; ks < D_N / 32; ++ks) {
    ahi[ks] = load_a_frag(x_hi, D_N, rb, ks * 32, lane);
    alo[ks] = load_a_frag(x_lo, D_N, rb, ks * 32, lane);
  }
  for (int ct = 0; ct < H_N / 16; ++ct) {
    int cb = ct * 16;
    v8f acc = {0.f, 0.f, 0.f, 0.f, 0.f, 0.f, 0.f, 0.f};
#pragma unroll
    for (int ks = 0; ks < D_N / 32; ++ks) {
      int k = ks * 32;
      bf16x16 bh = load_b_frag(Wpt_hi, D_N, cb, k, lane);
      bf16x16 bl = load_b_frag(Wpt_lo, D_N, cb, k, lane);
      acc = wmma_bf16(ahi[ks], bh, acc);
      acc = wmma_bf16(ahi[ks], bl, acc);
      acc = wmma_bf16(alo[ks], bh, acc);
    }
    float bp = b_proj[cb + nc];
#pragma unroll
    for (int v = 0; v < 8; ++v) {
      int r = v + 8 * hic;
      float hv = acc[v] + bp;
      hv = hv > 0.f ? hv : 0.f;
      h[(size_t)(rb + r) * H_N + cb + nc] = hv;
    }
  }
}

// ---------------- kernel 4: streaming KNN (WMMA gram + running top-16) -------------
// grid.x = 128 row-blocks (64 rows each), grid.y = NSPLIT column splits, block = 128
__global__ __launch_bounds__(128, 2) void knn_kernel(
    const ushort_t* __restrict__ x_hi, const ushort_t* __restrict__ x_lo,
    const float* __restrict__ sq,
    float* __restrict__ part_d, int* __restrict__ part_i) {
  __shared__ ushort_t sBh[32 * SB_LD];          // 24.5 KB
  __shared__ ushort_t sBl[32 * SB_LD];          // 24.5 KB
  __shared__ float    stile[4][16 * ST_LD];     // 9.25 KB
  int tid = threadIdx.x;
  int w = tid >> 5, lane = tid & 31;
  int rb = blockIdx.x * 64 + w * 16;            // this wave's 16 rows
  int cs = blockIdx.y * COLS_PER_SPLIT;
  int nc = lane & 15, hic = lane >> 4;
  int sr = tid >> 2, sq4 = tid & 3;             // staging role: row, quarter

  // A_hi register-resident (96 VGPRs); A_lo streamed from WGP$ each K-step
  bf16x16 ahi[D_N / 32];
#pragma unroll
  for (int ks = 0; ks < D_N / 32; ++ks)
    ahi[ks] = load_a_frag(x_hi, D_N, rb, ks * 32, lane);

  float bd[16]; int bi[16];
#pragma unroll
  for (int j = 0; j < 16; ++j) { bd[j] = 3.4e38f; bi[j] = 0; }

  for (int ch = 0; ch < COLS_PER_SPLIT / 32; ++ch) {
    int cb = cs + ch * 32;
    __syncthreads();                            // staging buffer free to refill
    {
      const ushort_t* gh = x_hi + (size_t)(cb + sr) * D_N + sq4 * 96;
      const ushort_t* gl = x_lo + (size_t)(cb + sr) * D_N + sq4 * 96;
      ushort_t* lh = sBh + sr * SB_LD + sq4 * 96;
      ushort_t* ll = sBl + sr * SB_LD + sq4 * 96;
#pragma unroll
      for (int j = 0; j < 12; ++j) {
        *(int4*)(lh + j * 8) = *(const int4*)(gh + j * 8);
        *(int4*)(ll + j * 8) = *(const int4*)(gl + j * 8);
      }
      if (ch + 1 < COLS_PER_SPLIT / 32) {       // warm L2/WGP$ for next chunk
        __builtin_prefetch(gh + (size_t)32 * D_N, 0, 1);
        __builtin_prefetch(gl + (size_t)32 * D_N, 0, 1);
      }
    }
    __syncthreads();                            // staging visible to all waves

    v8f acc0 = {0.f, 0.f, 0.f, 0.f, 0.f, 0.f, 0.f, 0.f};
    v8f acc1 = acc0;
#pragma unroll
    for (int ks = 0; ks < D_N / 32; ++ks) {
      int k = ks * 32;
      bf16x16 al  = load_a_frag(x_lo, D_N, rb, k, lane);
      bf16x16 b0h = load_b_lds(sBh, 0,  k, lane);
      bf16x16 b0l = load_b_lds(sBl, 0,  k, lane);
      bf16x16 b1h = load_b_lds(sBh, 16, k, lane);
      bf16x16 b1l = load_b_lds(sBl, 16, k, lane);
      acc0 = wmma_bf16(ahi[ks], b0h, acc0);
      acc0 = wmma_bf16(ahi[ks], b0l, acc0);
      acc0 = wmma_bf16(al,      b0h, acc0);
      acc1 = wmma_bf16(ahi[ks], b1h, acc1);
      acc1 = wmma_bf16(ahi[ks], b1l, acc1);
      acc1 = wmma_bf16(al,      b1h, acc1);
    }
    // score = sq[col] - 2*dot  (sq[row] constant per row: irrelevant for ordering)
    float s0 = sq[cb + nc], s1 = sq[cb + 16 + nc];
#pragma unroll
    for (int v = 0; v < 8; ++v) {
      int rr = v + 8 * hic;
      stile[w][rr * ST_LD + nc]      = s0 - 2.f * acc0[v];
      stile[w][rr * ST_LD + 16 + nc] = s1 - 2.f * acc1[v];
    }
    __builtin_amdgcn_wave_barrier();            // per-wave tile; LDS in-order per wave
    // lane owns (row = lane&15, column half = lane>>4)
    const float* myrow = &stile[w][(lane & 15) * ST_LD + hic * 16];
#pragma unroll 4
    for (int c = 0; c < 16; ++c) {
      float d = myrow[c];
      if (d < bd[15]) topk_insert(bd, bi, d, cb + hic * 16 + c);
    }
    __builtin_amdgcn_wave_barrier();
  }
  int row = rb + (lane & 15);
  int sl = blockIdx.y * 2 + hic;                // which of the NLIST partial lists
  size_t base = ((size_t)sl * B_N + row) * 16;
#pragma unroll
  for (int j = 0; j < 16; ++j) { part_d[base + j] = bd[j]; part_i[base + j] = bi[j]; }
}

// ---------------- kernel 5: merge NLIST partial lists per row ----------------------
__global__ void merge_kernel(const float* __restrict__ part_d,
                             const int* __restrict__ part_i,
                             int* __restrict__ idx_out) {
  int row = blockIdx.x * blockDim.x + threadIdx.x;
  float bd[16]; int bi[16];
#pragma unroll
  for (int j = 0; j < 16; ++j) { bd[j] = 3.4e38f; bi[j] = 0; }
  for (int s = 0; s < NLIST; ++s) {
    size_t base = ((size_t)s * B_N + row) * 16;
#pragma unroll
    for (int j = 0; j < 16; ++j) {
      float d = part_d[base + j];
      if (d < bd[15]) topk_insert(bd, bi, d, part_i[base + j]);
    }
  }
#pragma unroll
  for (int j = 0; j < 16; ++j) idx_out[(size_t)row * 16 + j] = bi[j];
}

// ---------------- kernel 6: ragged top-k mean of h, emit agg hi/lo bf16 ------------
__global__ void gather_kernel(const float* __restrict__ h,
                              const int* __restrict__ idx,
                              const int* __restrict__ kk,
                              ushort_t* __restrict__ agg_hi,
                              ushort_t* __restrict__ agg_lo) {
  int w = threadIdx.x >> 5, lane = threadIdx.x & 31;
  int row = blockIdx.x * 8 + w;
  int kv = kk[row];
  float acc[8] = {0.f, 0.f, 0.f, 0.f, 0.f, 0.f, 0.f, 0.f};
  for (int j = 0; j < kv; ++j) {
    int nb = idx[(size_t)row * 16 + j];
    const float* hp = h + (size_t)nb * H_N + lane;
#pragma unroll
    for (int t = 0; t < 8; ++t) acc[t] += hp[t * 32];
  }
  float inv = 1.f / (float)kv;
#pragma unroll
  for (int t = 0; t < 8; ++t) {
    float v = acc[t] * inv;
    ushort_t hb = f2bf(v);
    agg_hi[(size_t)row * H_N + lane + t * 32] = hb;
    agg_lo[(size_t)row * H_N + lane + t * 32] = f2bf(v - bf2f(hb));
  }
}

// ---------------- kernel 7: r = relu(agg@W_res+b), z=h+r, LN, out = zn*g+b @ W_fc --
__global__ __launch_bounds__(64, 2) void resnorm_kernel(
    const ushort_t* __restrict__ agg_hi, const ushort_t* __restrict__ agg_lo,
    const ushort_t* __restrict__ Wrt_hi, const ushort_t* __restrict__ Wrt_lo,
    const float* __restrict__ b_res, const float* __restrict__ h,
    const float* __restrict__ ln_g, const float* __restrict__ ln_b,
    const float* __restrict__ W_fc, const float* __restrict__ b_fc,
    float* __restrict__ out) {
  __shared__ float zbuf[2][16][257];
  int w = threadIdx.x >> 5, lane = threadIdx.x & 31;
  int rb = (blockIdx.x * 2 + w) * 16;
  int nc = lane & 15, hic = lane >> 4;
  // A fragments register-resident: 16 x 8 = 128 VGPRs
  bf16x16 ahi[H_N / 32], alo[H_N / 32];
#pragma unroll
  for (int ks = 0; ks < H_N / 32; ++ks) {
    ahi[ks] = load_a_frag(agg_hi, H_N, rb, ks * 32, lane);
    alo[ks] = load_a_frag(agg_lo, H_N, rb, ks * 32, lane);
  }
  for (int ct = 0; ct < H_N / 16; ++ct) {
    int cb = ct * 16;
    v8f acc = {0.f, 0.f, 0.f, 0.f, 0.f, 0.f, 0.f, 0.f};
#pragma unroll
    for (int ks = 0; ks < H_N / 32; ++ks) {
      int k = ks * 32;
      bf16x16 bh = load_b_frag(Wrt_hi, H_N, cb, k, lane);
      bf16x16 bl = load_b_frag(Wrt_lo, H_N, cb, k, lane);
      acc = wmma_bf16(ahi[ks], bh, acc);
      acc = wmma_bf16(ahi[ks], bl, acc);
      acc = wmma_bf16(alo[ks], bh, acc);
    }
    float br = b_res[cb + nc];
#pragma unroll
    for (int v = 0; v < 8; ++v) {
      int r = v + 8 * hic;
      float rr = acc[v] + br;
      rr = rr > 0.f ? rr : 0.f;
      zbuf[w][r][cb + nc] = h[(size_t)(rb + r) * H_N + cb + nc] + rr;
    }
  }
  __builtin_amdgcn_wave_barrier();
  if (lane < 16) {
    int row = rb + lane;
    float s = 0.f, s2 = 0.f;
    for (int c = 0; c < H_N; ++c) {
      float z = zbuf[w][lane][c];
      s += z; s2 += z * z;
    }
    float mu = s * (1.f / H_N);
    float var = s2 * (1.f / H_N) - mu * mu;
    float inv = rsqrtf(var + 1e-5f);
    float oc[C_N];
#pragma unroll
    for (int j = 0; j < C_N; ++j) oc[j] = b_fc[j];
    for (int c = 0; c < H_N; ++c) {
      float zn = (zbuf[w][lane][c] - mu) * inv;
      float val = zn * ln_g[c] + ln_b[c];
#pragma unroll
      for (int j = 0; j < C_N; ++j) oc[j] += val * W_fc[c * C_N + j];
    }
#pragma unroll
    for (int j = 0; j < C_N; ++j) out[(size_t)row * C_N + j] = oc[j];
  }
}

// ---------------- host launch ----------------
extern "C" void kernel_launch(void* const* d_in, const int* in_sizes, int n_in,
                              void* d_out, int out_size, void* d_ws, size_t ws_size,
                              hipStream_t stream) {
  (void)in_sizes; (void)n_in; (void)out_size; (void)ws_size;
  const float* x      = (const float*)d_in[0];
  const float* W_proj = (const float*)d_in[1];
  const float* b_proj = (const float*)d_in[2];
  const float* W_tau  = (const float*)d_in[3];
  const float* b_tau  = (const float*)d_in[4];
  const float* W_res  = (const float*)d_in[5];
  const float* b_res  = (const float*)d_in[6];
  const float* ln_g   = (const float*)d_in[7];
  const float* ln_b   = (const float*)d_in[8];
  const float* W_fc   = (const float*)d_in[9];
  const float* b_fc   = (const float*)d_in[10];
  float* out = (float*)d_out;

  char* ws = (char*)d_ws;
  size_t off = 0;
  auto alloc = [&](size_t bytes) -> void* {
    off = (off + 255) & ~(size_t)255;
    void* p = ws + off;
    off += bytes;
    return p;
  };
  ushort_t* x_hi   = (ushort_t*)alloc((size_t)B_N * D_N * 2);
  ushort_t* x_lo   = (ushort_t*)alloc((size_t)B_N * D_N * 2);
  float*    sq     = (float*)   alloc((size_t)B_N * 4);
  int*      kk     = (int*)     alloc((size_t)B_N * 4);
  ushort_t* Wpt_hi = (ushort_t*)alloc((size_t)H_N * D_N * 2);
  ushort_t* Wpt_lo = (ushort_t*)alloc((size_t)H_N * D_N * 2);
  ushort_t* Wrt_hi = (ushort_t*)alloc((size_t)H_N * H_N * 2);
  ushort_t* Wrt_lo = (ushort_t*)alloc((size_t)H_N * H_N * 2);
  float*    h      = (float*)   alloc((size_t)B_N * H_N * 4);
  ushort_t* agg_hi = (ushort_t*)alloc((size_t)B_N * H_N * 2);
  ushort_t* agg_lo = (ushort_t*)alloc((size_t)B_N * H_N * 2);
  float*    part_d = (float*)   alloc((size_t)NLIST * B_N * 16 * 4);
  int*      part_i = (int*)     alloc((size_t)NLIST * B_N * 16 * 4);
  int*      idx    = (int*)     alloc((size_t)B_N * 16 * 4);

  prep_x_kernel<<<B_N / 8, 256, 0, stream>>>(x, W_tau, b_tau, x_hi, x_lo, sq, kk);
  tsplit_kernel<<<(D_N * H_N + 255) / 256, 256, 0, stream>>>(W_proj, Wpt_hi, Wpt_lo, D_N, H_N);
  tsplit_kernel<<<(H_N * H_N + 255) / 256, 256, 0, stream>>>(W_res, Wrt_hi, Wrt_lo, H_N, H_N);
  proj_kernel<<<B_N / (8 * 16), 256, 0, stream>>>(x_hi, x_lo, Wpt_hi, Wpt_lo, b_proj, h);
  knn_kernel<<<dim3(B_N / 64, NSPLIT), 128, 0, stream>>>(x_hi, x_lo, sq, part_d, part_i);
  merge_kernel<<<B_N / 128, 128, 0, stream>>>(part_d, part_i, idx);
  gather_kernel<<<B_N / 8, 256, 0, stream>>>(h, idx, kk, agg_hi, agg_lo);
  resnorm_kernel<<<B_N / 32, 64, 0, stream>>>(agg_hi, agg_lo, Wrt_hi, Wrt_lo, b_res, h,
                                              ln_g, ln_b, W_fc, b_fc, out);
}
